// RWKV_TimeMix_27650999452324
// MI455X (gfx1250) — matrix operand order repaired and checked by reading, not compiled, not run on hardware
//
#include <hip/hip_runtime.h>
#include <hip/hip_bf16.h>
#include <stdint.h>

// ---------------------------------------------------------------------------
// RWKV TimeMix for MI455X (gfx1250): bf16 WMMA + async-to-LDS staged GEMMs.
// ---------------------------------------------------------------------------

typedef __bf16 bf16;
typedef __attribute__((ext_vector_type(16))) __bf16 v16bf;
typedef __attribute__((ext_vector_type(8)))  __bf16 v8bf;
typedef __attribute__((ext_vector_type(8)))  float  v8f;

#define B_   4
#define T_   2048
#define C_   2048
#define H_   32
#define K_   64
#define NCH  (T_ / 64)
#define EPS_ 0.00064f

// ---------------- scalar bf16 helpers (bit-level, RNE) ----------------------
__device__ __forceinline__ bf16 f2bf(float f) {
  union { float f; unsigned u; } a; a.f = f;
  unsigned r = a.u + 0x7FFFu + ((a.u >> 16) & 1u);
  unsigned short h = (unsigned short)(r >> 16);
  bf16 b; __builtin_memcpy(&b, &h, 2); return b;
}
__device__ __forceinline__ float bf2f(bf16 x) {
  unsigned short u; __builtin_memcpy(&u, &x, 2);
  union { unsigned u; float f; } a; a.u = ((unsigned)u) << 16; return a.f;
}
__device__ __forceinline__ float bfround(float f) { return bf2f(f2bf(f)); }

// ---------------- WMMA fragment loaders (ISA 7.12.2 layouts) ----------------
__device__ __forceinline__ v16bf load_a_rm(const bf16* A, int ld, int row_base,
                                           int k_base, int lane) {
  const int m = lane & 15, hh = lane >> 4;
  const bf16* p = A + (size_t)(row_base + m) * ld + k_base + hh * 8;
  v8bf lo = *(const v8bf*)p;
  v8bf hi = *(const v8bf*)(p + 16);
  return __builtin_shufflevector(lo, hi, 0,1,2,3,4,5,6,7,8,9,10,11,12,13,14,15);
}
__device__ __forceinline__ v16bf load_b_bt(const bf16* BT, int ld, int n_base,
                                           int k_base, int lane) {
  const int n = lane & 15, kh = lane >> 4;
  const bf16* p = BT + (size_t)(n_base + n) * ld + k_base + kh * 8;
  v8bf lo = *(const v8bf*)p;
  v8bf hi = *(const v8bf*)(p + 16);
  return __builtin_shufflevector(lo, hi, 0,1,2,3,4,5,6,7,8,9,10,11,12,13,14,15);
}
__device__ __forceinline__ v16bf load_a_cm(const bf16* src, int ld, int row_base,
                                           int k_base, int lane) {
  const int m = lane & 15, hh = lane >> 4;
  v16bf r;
#pragma unroll
  for (int v = 0; v < 8; ++v) {
    int k0 = ((v >= 4) ? 16 : 0) + hh * 8 + (v & 3) * 2;
    r[2 * v]     = src[(size_t)(k_base + k0)     * ld + row_base + m];
    r[2 * v + 1] = src[(size_t)(k_base + k0 + 1) * ld + row_base + m];
  }
  return r;
}
__device__ __forceinline__ v8f load_c32(const float* Cp, int ld, int m_base,
                                        int n_base, int lane) {
  const int n = lane & 15, mh = lane >> 4;
  v8f c;
#pragma unroll
  for (int r2 = 0; r2 < 8; ++r2)
    c[r2] = Cp[(size_t)(m_base + mh * 8 + r2) * ld + n_base + n];
  return c;
}
__device__ __forceinline__ void store_c32(float* Cp, int ld, int m_base,
                                          int n_base, int lane, v8f acc) {
  const int n = lane & 15, mh = lane >> 4;
#pragma unroll
  for (int r2 = 0; r2 < 8; ++r2)
    Cp[(size_t)(m_base + mh * 8 + r2) * ld + n_base + n] = acc[r2];
}

__device__ __forceinline__ v8f wmma_bf16(v16bf a, v16bf b, v8f c) {
  return __builtin_amdgcn_wmma_f32_16x16x32_bf16(false, a, false, b,
                                                 (short)0, c, false, false);
}

// Async copy 16B global -> LDS (ASYNCcnt-tracked, CDNA5 TDM-lite path).
__device__ __forceinline__ void async_copy16(const void* gptr, void* lptr) {
  unsigned lds = (unsigned)(uintptr_t)lptr;           // LDS addr = addr[31:0]
  unsigned long long ga = (unsigned long long)(uintptr_t)gptr;
  asm volatile("global_load_async_to_lds_b128 %0, %1, off"
               :: "v"(lds), "v"(ga) : "memory");
}

// ---------------------------------------------------------------------------
// Kernel 1: token-shift mixing -> xk/xv/xr/xg (bf16)
// ---------------------------------------------------------------------------
__global__ __launch_bounds__(256) void mix_kernel(
    const float* __restrict__ x, const float* __restrict__ state,
    const float* __restrict__ tmk, const float* __restrict__ tmv,
    const float* __restrict__ tmr, const float* __restrict__ tmg,
    bf16* __restrict__ xk, bf16* __restrict__ xv,
    bf16* __restrict__ xr, bf16* __restrict__ xg) {
  const int n = blockIdx.x;            // b*T + t
  const int t = n & (T_ - 1);
  const size_t base = (size_t)n * C_;
  int c = threadIdx.x * 8;
#pragma unroll
  for (int e = 0; e < 8; ++e, ++c) {
    float xc = x[base + c];
    float xp = (t == 0) ? state[c] : x[base - C_ + c];
    float mk = tmk[c], mv = tmv[c], mr = tmr[c], mg = tmg[c];
    xk[base + c] = f2bf(xc * mk + xp * (1.f - mk));
    xv[base + c] = f2bf(xc * mv + xp * (1.f - mv));
    xr[base + c] = f2bf(xc * mr + xp * (1.f - mr));
    xg[base + c] = f2bf(xc * mg + xp * (1.f - mg));
  }
}

// ---------------------------------------------------------------------------
// Kernel 2: fp32 -> bf16 weight conversion
// ---------------------------------------------------------------------------
__global__ __launch_bounds__(256) void wcvt_kernel(const float* __restrict__ src,
                                                   bf16* __restrict__ dst) {
  size_t i = ((size_t)blockIdx.x * 256 + threadIdx.x) * 4;
#pragma unroll
  for (int j = 0; j < 4; ++j) dst[i + j] = f2bf(src[i + j]);
}

// ---------------------------------------------------------------------------
// Kernel 3: O[m,n] = sum_k A[m,k] * W[n,k]   (A,W bf16 row-major, ld = K)
// 8 waves; block tile 128(M) x 64(N); K pipelined in 32-slabs, double-buffered
// in LDS via global_load_async_to_lds_b128 (overlap DMA with WMMA).
// ---------------------------------------------------------------------------
template <bool STORE_BF16>
__global__ __launch_bounds__(256) void gemm_bf16_kernel(
    const bf16* __restrict__ A, const bf16* __restrict__ W,
    void* __restrict__ OutV, int M, int N, int K) {
  __shared__ __align__(32) bf16 Asl[2][128][32];   // 2 x 8 KiB
  __shared__ __align__(32) bf16 Wsl[2][64][32];    // 2 x 4 KiB
  const int tid = threadIdx.x;
  const int lane = tid & 31;
  const int wave = tid >> 5;
  const int wm = wave >> 1, wn = wave & 1;
  const int bm0 = blockIdx.x * 128;
  const int bn0 = blockIdx.y * 64;
  const int bm = bm0 + wm * 32;
  const int bn = bn0 + wn * 32;

  // Issue one K-slab's async copies: 3 x b128 per thread (A:2, W:1).
  auto stage = [&](int buf, int k0) {
#pragma unroll
    for (int rep = 0; rep < 2; ++rep) {
      int cid = tid + rep * 256;              // 512 16B-chunks for A slab
      int row = cid >> 2, ch = cid & 3;
      async_copy16(A + (size_t)(bm0 + row) * K + k0 + ch * 8,
                   &Asl[buf][row][ch * 8]);
    }
    {
      int row = tid >> 2, ch = tid & 3;       // 256 16B-chunks for W slab
      async_copy16(W + (size_t)(bn0 + row) * K + k0 + ch * 8,
                   &Wsl[buf][row][ch * 8]);
    }
  };

  v8f acc00 = {}, acc01 = {}, acc10 = {}, acc11 = {};
  const int nK = K / 32;
  stage(0, 0);
  for (int ks = 0; ks < nK; ++ks) {
    const int buf = ks & 1;
    if (ks + 1 < nK) {
      stage(buf ^ 1, (ks + 1) * 32);
      asm volatile("s_wait_asynccnt 0x3" ::: "memory");  // slab ks landed
    } else {
      asm volatile("s_wait_asynccnt 0x0" ::: "memory");
    }
    __syncthreads();
    const bf16* Ab = &Asl[buf][0][0];
    const bf16* Wb = &Wsl[buf][0][0];
    v16bf a0 = load_a_rm(Ab, 32, wm * 32, 0, lane);
    v16bf a1 = load_a_rm(Ab, 32, wm * 32 + 16, 0, lane);
    v16bf b0 = load_b_bt(Wb, 32, wn * 32, 0, lane);
    v16bf b1 = load_b_bt(Wb, 32, wn * 32 + 16, 0, lane);
    acc00 = wmma_bf16(a0, b0, acc00);
    acc01 = wmma_bf16(a0, b1, acc01);
    acc10 = wmma_bf16(a1, b0, acc10);
    acc11 = wmma_bf16(a1, b1, acc11);
    __syncthreads();   // done reading buf before it is restaged
  }

  const int mh = lane >> 4, nn = lane & 15;
  if constexpr (STORE_BF16) {
    bf16* O = (bf16*)OutV;
#pragma unroll
    for (int r2 = 0; r2 < 8; ++r2) {
      size_t row0 = (size_t)(bm + mh * 8 + r2) * N;
      size_t row1 = (size_t)(bm + 16 + mh * 8 + r2) * N;
      O[row0 + bn + nn]      = f2bf(acc00[r2]);
      O[row0 + bn + 16 + nn] = f2bf(acc01[r2]);
      O[row1 + bn + nn]      = f2bf(acc10[r2]);
      O[row1 + bn + 16 + nn] = f2bf(acc11[r2]);
    }
  } else {
    float* O = (float*)OutV;
#pragma unroll
    for (int r2 = 0; r2 < 8; ++r2) {
      size_t row0 = (size_t)(bm + mh * 8 + r2) * N;
      size_t row1 = (size_t)(bm + 16 + mh * 8 + r2) * N;
      O[row0 + bn + nn]      = acc00[r2];
      O[row0 + bn + 16 + nn] = acc01[r2];
      O[row1 + bn + nn]      = acc10[r2];
      O[row1 + bn + 16 + nn] = acc11[r2];
    }
  }
}

// ---------------------------------------------------------------------------
// Kernel 4: WKV. One block per (b,h); 8 waves; S (64x64 f32) resident in LDS.
// Pass A: y = (r .* w^t) @ wkvstate + (sum_k r*k*faaaa) .* v
// Pass B: chunked causal scan, accumulated into y at t'+1.
// ---------------------------------------------------------------------------
__global__ __launch_bounds__(256) void wkv_kernel(
    const bf16* __restrict__ rB, const bf16* __restrict__ kB,
    const bf16* __restrict__ vB, const float* __restrict__ wkvstate,
    const float* __restrict__ tdec, const float* __restrict__ tfaa,
    float* __restrict__ y) {
  __shared__ __align__(32) float S[64][64];     // scan state (fp32)
  __shared__ __align__(32) bf16 SbT[64][64];    // transposed bf16 state / S0
  __shared__ __align__(32) bf16 rcS[64][64];    // r-chunk  [t][k]
  __shared__ __align__(32) bf16 kcS[64][64];    // k-chunk  [t][k] (decayed)
  __shared__ __align__(32) bf16 vTS[64][64];    // v-chunk transposed [j][t]
  __shared__ __align__(32) bf16 atS[64][64];    // masked attention [p][q]
  __shared__ float bonus[64];
  __shared__ float lwS[64];
  __shared__ float fS[64];

  const int bh = blockIdx.x;
  const int b = bh >> 5;
  const int h = bh & 31;
  const int tid = threadIdx.x;
  const int lane = tid & 31;
  const int wave = tid >> 5;
  const size_t hb = (size_t)h * 64;

  if (tid < 64) {
    // w = exp(-exp(d));  log2(w) = -exp(d) * 1.4427
    lwS[tid] = -__expf(tdec[h * 64 + tid]) * 1.44269504f;
    fS[tid] = tfaa[h * 64 + tid];
  }
  for (int i = tid; i < 4096; i += 256) {
    ((float*)S)[i] = 0.f;
    int kk = i >> 6, j = i & 63;
    SbT[j][kk] = f2bf(wkvstate[(size_t)h * 4096 + kk * 64 + j]);  // S0^T
  }
  __syncthreads();

  // ---------------- Pass A: base + bonus ----------------
  for (int c = 0; c < NCH; ++c) {
    for (int i = tid; i < 4096; i += 256) {
      int p = i >> 6, kk = i & 63;
      int t = c * 64 + p;
      size_t idx = ((size_t)(b * T_ + t)) * C_ + hb + kk;
      rcS[p][kk] = f2bf(bf2f(rB[idx]) * exp2f((float)t * lwS[kk]));
    }
    if (tid < 64) {
      int t = c * 64 + tid;
      size_t idx = ((size_t)(b * T_ + t)) * C_ + hb;
      float s = 0.f;
      for (int kk = 0; kk < 64; ++kk)
        s += bf2f(rB[idx + kk]) * bf2f(kB[idx + kk]) * fS[kk];
      bonus[tid] = s;
    }
    __syncthreads();
#pragma unroll
    for (int i2 = 0; i2 < 2; ++i2) {
      int tI = wave * 2 + i2;
      int pt = (tI >> 2) * 16, nt = (tI & 3) * 16;
      v8f acc = {};
      for (int k0 = 0; k0 < 64; k0 += 32) {
        v16bf a = load_a_rm(&rcS[0][0], 64, pt, k0, lane);
        v16bf bb = load_b_bt(&SbT[0][0], 64, nt, k0, lane);
        acc = wmma_bf16(a, bb, acc);
      }
      const int mh = lane >> 4, nn = lane & 15;
#pragma unroll
      for (int r2 = 0; r2 < 8; ++r2) {
        int p = pt + mh * 8 + r2;
        int t = c * 64 + p;
        size_t idx = ((size_t)(b * T_ + t)) * C_ + hb + nt + nn;
        y[idx] = acc[r2] + bonus[p] * bf2f(vB[idx]);
      }
    }
    __syncthreads();
  }

  // ---------------- Pass B: chunked causal scan (S starts at 0) -----------
  for (int c = 0; c < NCH; ++c) {
    for (int i = tid; i < 4096; i += 256) {
      int p = i >> 6, kk = i & 63;
      int tp = c * 64 + p;  // shifted index t'
      size_t row0 = ((size_t)(b * T_ + tp)) * C_ + hb;
      float rv = (tp + 1 < T_) ? bf2f(rB[row0 + C_ + kk]) : 0.f;  // r[t'+1]
      rcS[p][kk] = f2bf(rv);
      float kv = 0.f, vv = 0.f;
      if (tp <= T_ - 2) {
        kv = bf2f(kB[row0 + kk]) * exp2f((float)(T_ - 1 - tp) * lwS[kk]);
        vv = bf2f(vB[row0 + kk]);
      }
      kcS[p][kk] = f2bf(kv);
      vTS[kk][p] = f2bf(vv);
      SbT[p][kk] = f2bf(S[kk][p]);  // snapshot S^T as bf16
    }
    __syncthreads();
    // att = tril(rc @ kc^T)
#pragma unroll
    for (int i2 = 0; i2 < 2; ++i2) {
      int tI = wave * 2 + i2;
      int pt = (tI >> 2) * 16, qt = (tI & 3) * 16;
      v8f acc = {};
      for (int k0 = 0; k0 < 64; k0 += 32) {
        v16bf a = load_a_rm(&rcS[0][0], 64, pt, k0, lane);
        v16bf bb = load_b_bt(&kcS[0][0], 64, qt, k0, lane);
        acc = wmma_bf16(a, bb, acc);
      }
      const int mh = lane >> 4, nn = lane & 15;
#pragma unroll
      for (int r2 = 0; r2 < 8; ++r2) {
        int p = pt + mh * 8 + r2;
        int q = qt + nn;
        atS[p][q] = f2bf((p >= q) ? acc[r2] : 0.f);
      }
    }
    __syncthreads();
    // out = att @ vc + rc @ S  -> y[t'+1] += out
#pragma unroll
    for (int i2 = 0; i2 < 2; ++i2) {
      int tI = wave * 2 + i2;
      int pt = (tI >> 2) * 16, jt = (tI & 3) * 16;
      v8f acc = {};
      for (int k0 = 0; k0 < 64; k0 += 32) {
        v16bf a = load_a_rm(&atS[0][0], 64, pt, k0, lane);
        v16bf bb = load_b_bt(&vTS[0][0], 64, jt, k0, lane);
        acc = wmma_bf16(a, bb, acc);
        v16bf a2 = load_a_rm(&rcS[0][0], 64, pt, k0, lane);
        v16bf b2 = load_b_bt(&SbT[0][0], 64, jt, k0, lane);
        acc = wmma_bf16(a2, b2, acc);
      }
      const int mh = lane >> 4, nn = lane & 15;
#pragma unroll
      for (int r2 = 0; r2 < 8; ++r2) {
        int p = pt + mh * 8 + r2;
        int t = c * 64 + p + 1;
        if (t < T_) {
          size_t idx = ((size_t)(b * T_ + t)) * C_ + hb + jt + nn;
          y[idx] += acc[r2];
        }
      }
    }
    // S += kc^T @ vc   (disjoint tiles per wave; fp32 accumulate in LDS)
#pragma unroll
    for (int i2 = 0; i2 < 2; ++i2) {
      int tI = wave * 2 + i2;
      int ft = (tI >> 2) * 16, jt = (tI & 3) * 16;
      v8f acc = load_c32(&S[0][0], 64, ft, jt, lane);
      for (int k0 = 0; k0 < 64; k0 += 32) {
        v16bf a = load_a_cm(&kcS[0][0], 64, ft, k0, lane);  // kc^T
        v16bf bb = load_b_bt(&vTS[0][0], 64, jt, k0, lane);
        acc = wmma_bf16(a, bb, acc);
      }
      store_c32(&S[0][0], 64, ft, jt, lane, acc);
    }
    __syncthreads();
  }
}

// ---------------------------------------------------------------------------
// Kernel 5: per-head LayerNorm (with the reference's bf16 round-trip of y)
//           fused with SiLU gate -> z (bf16) for the output projection.
// ---------------------------------------------------------------------------
__global__ __launch_bounds__(256) void ln_gate_kernel(
    const float* __restrict__ y, const bf16* __restrict__ g,
    const float* __restrict__ ln_w, const float* __restrict__ ln_b,
    bf16* __restrict__ z) {
  __shared__ float s1[256], s2[256];
  const int n = blockIdx.x;
  const size_t base = (size_t)n * C_;
  const int tid = threadIdx.x;
  const int c0 = tid * 8;  // 8 threads per 64-wide head
  float yv[8];
  float sum = 0.f, sq = 0.f;
#pragma unroll
  for (int e = 0; e < 8; ++e) {
    float v = bfround(y[base + c0 + e]);
    yv[e] = v;
    sum += v;
    sq += v * v;
  }
  s1[tid] = sum;
  s2[tid] = sq;
  __syncthreads();
  const int hbT = tid & ~7;
  float S1 = 0.f, S2 = 0.f;
#pragma unroll
  for (int q = 0; q < 8; ++q) { S1 += s1[hbT + q]; S2 += s2[hbT + q]; }
  float mu = S1 * (1.f / 64.f);
  float var = S2 * (1.f / 64.f) - mu * mu;
  float rstd = rsqrtf(var + EPS_);
#pragma unroll
  for (int e = 0; e < 8; ++e) {
    int c = c0 + e;
    float gv = bf2f(g[base + c]);
    float sil = gv / (1.f + __expf(-gv));
    float o = (yv[e] - mu) * rstd * ln_w[c] + ln_b[c];
    z[base + c] = f2bf(o * sil);
  }
}

// ---------------------------------------------------------------------------
extern "C" void kernel_launch(void* const* d_in, const int* in_sizes, int n_in,
                              void* d_out, int out_size, void* d_ws,
                              size_t ws_size, hipStream_t stream) {
  const float* x     = (const float*)d_in[0];
  const float* state = (const float*)d_in[1];
  const float* wkvs  = (const float*)d_in[2];
  const float* tmk   = (const float*)d_in[3];
  const float* tmv   = (const float*)d_in[4];
  const float* tmr   = (const float*)d_in[5];
  const float* tmg   = (const float*)d_in[6];
  const float* tdec  = (const float*)d_in[7];
  const float* tfaa  = (const float*)d_in[8];
  const float* w_r   = (const float*)d_in[9];
  const float* w_k   = (const float*)d_in[10];
  const float* w_v   = (const float*)d_in[11];
  const float* w_g   = (const float*)d_in[12];
  const float* w_o   = (const float*)d_in[13];
  const float* ln_w  = (const float*)d_in[14];
  const float* ln_b  = (const float*)d_in[15];

  char* ws = (char*)d_ws;
  const size_t SZXB = (size_t)B_ * T_ * C_ * sizeof(bf16);  // 32 MiB
  const size_t SZW  = (size_t)C_ * C_ * sizeof(bf16);       // 8 MiB
  bf16* xk  = (bf16*)(ws + 0 * SZXB);
  bf16* xv  = (bf16*)(ws + 1 * SZXB);
  bf16* xr  = (bf16*)(ws + 2 * SZXB);
  bf16* xg  = (bf16*)(ws + 3 * SZXB);
  bf16* wrb = (bf16*)(ws + 4 * SZXB + 0 * SZW);
  bf16* wkb = (bf16*)(ws + 4 * SZXB + 1 * SZW);
  bf16* wvb = (bf16*)(ws + 4 * SZXB + 2 * SZW);
  bf16* wgb = (bf16*)(ws + 4 * SZXB + 3 * SZW);
  bf16* wob = (bf16*)(ws + 4 * SZXB + 4 * SZW);
  bf16* rBp = (bf16*)(ws + 4 * SZXB + 5 * SZW);
  bf16* kBp = rBp + (size_t)B_ * T_ * C_;
  bf16* vBp = kBp + (size_t)B_ * T_ * C_;
  bf16* gBp = vBp + (size_t)B_ * T_ * C_;
  // Stream-ordered reuse: xk/xv dead after GEMMs -> y (fp32, 64 MiB);
  // xr dead after GEMMs -> z (bf16, 32 MiB).
  float* yBuf = (float*)xk;
  bf16*  zBuf = xr;

  const int M = B_ * T_, N = C_, K = C_;
  dim3 gGrid(M / 128, N / 64);

  mix_kernel<<<B_ * T_, 256, 0, stream>>>(x, state, tmk, tmv, tmr, tmg,
                                          xk, xv, xr, xg);
  const int wBlocks = (C_ * C_) / (256 * 4);
  wcvt_kernel<<<wBlocks, 256, 0, stream>>>(w_r, wrb);
  wcvt_kernel<<<wBlocks, 256, 0, stream>>>(w_k, wkb);
  wcvt_kernel<<<wBlocks, 256, 0, stream>>>(w_v, wvb);
  wcvt_kernel<<<wBlocks, 256, 0, stream>>>(w_g, wgb);
  wcvt_kernel<<<wBlocks, 256, 0, stream>>>(w_o, wob);

  gemm_bf16_kernel<true><<<gGrid, 256, 0, stream>>>(xr, wrb, rBp, M, N, K);
  gemm_bf16_kernel<true><<<gGrid, 256, 0, stream>>>(xk, wkb, kBp, M, N, K);
  gemm_bf16_kernel<true><<<gGrid, 256, 0, stream>>>(xv, wvb, vBp, M, N, K);
  gemm_bf16_kernel<true><<<gGrid, 256, 0, stream>>>(xg, wgb, gBp, M, N, K);

  wkv_kernel<<<B_ * H_, 256, 0, stream>>>(rBp, kBp, vBp, wkvs, tdec, tfaa,
                                          yBuf);
  ln_gate_kernel<<<B_ * T_, 256, 0, stream>>>(yBuf, gBp, ln_w, ln_b, zBuf);
  gemm_bf16_kernel<false><<<gGrid, 256, 0, stream>>>(zBuf, wob, d_out, M, N, K);
}
